// GraphMertGraphEncoder_22024592294145
// MI455X (gfx1250) — compile-verified, matrix operand order checked by reading, not run on hardware
//
#include <hip/hip_runtime.h>

// ---------------------------------------------------------------------------
// GraphMert encoder, 6 layers, MI455X (gfx1250, wave32).
// GEMMs: v_wmma_f32_16x16x32_f16, B-tiles staged via global_load_async_to_lds
// (double-buffered LDS, ASYNCcnt), weights pre-converted to f16 W^T once.
// LayerNorm / softmax / GELU / residuals in fp32.
// ---------------------------------------------------------------------------

typedef __attribute__((ext_vector_type(16))) _Float16 v16h;
typedef __attribute__((ext_vector_type(8)))  float    v8f;

union Frag16 { v16h h; unsigned int u[8]; };

#define Lc 6
#define Hc 768
#define NHc 12
#define HDc 64
#define FFc 3072
#define Tc 512
#define Bc 4
#define Mtok 2048            // T*B tokens
#define NEG_INF (-__builtin_inff())

// k-offset inside a 16-bit A/B fragment (wave32, 16x16x32):
// dword j (covering elements 2j,2j+1), hsel = lane/16.
__device__ __forceinline__ int koff(int j, int hsel) {
  return (j < 4) ? (8 * hsel + 2 * j) : (16 + 8 * hsel + 2 * (j - 4));
}

__device__ __forceinline__ float gelu_f(float x) {
  const float c = 0.7978845608028654f;             // sqrt(2/pi)
  float t = c * (x + 0.044715f * x * x * x);
  return 0.5f * x * (1.0f + tanhf(t));
}

// CDNA5 async global->LDS copy (ASYNCcnt), 16 bytes per lane.
__device__ __forceinline__ void async_b128(unsigned lds, const _Float16* g) {
  asm volatile("global_load_async_to_lds_b128 %0, %1, off"
               :: "v"(lds), "v"(g) : "memory");
}
__device__ __forceinline__ void wait_async0() {
#if __has_builtin(__builtin_amdgcn_s_wait_asynccnt)
  __builtin_amdgcn_s_wait_asynccnt(0);
#else
  asm volatile("s_wait_asynccnt 0" ::: "memory");
#endif
}

// ---------------------------------------------------------------------------
// One-time weight prep: W(KxN, f32 row-major) -> W^T(NxK, f16 row-major),
// LDS-tiled 64x64 transpose so both global streams are coalesced.
// ---------------------------------------------------------------------------
__global__ __launch_bounds__(256) void transpose_w(
    const float* __restrict__ W, _Float16* __restrict__ WT, int K, int N) {
  __shared__ float tile[64][65];
  const int k0 = blockIdx.x * 64, n0 = blockIdx.y * 64;
  const int tid = threadIdx.x;
#pragma unroll
  for (int p = 0; p < 16; ++p) {
    int i = tid + p * 256;
    int kk = i >> 6, nn = i & 63;
    tile[kk][nn] = W[(size_t)(k0 + kk) * N + (n0 + nn)];
  }
  __syncthreads();
#pragma unroll
  for (int p = 0; p < 16; ++p) {
    int i = tid + p * 256;
    int nn = i >> 6, kk = i & 63;
    WT[(size_t)(n0 + nn) * K + (k0 + kk)] = (_Float16)tile[kk][nn];
  }
}

// ---------------------------------------------------------------------------
// LayerNorm: one block per token, H=768, 3 elements/thread. f32 in, f16 out.
// ---------------------------------------------------------------------------
__global__ __launch_bounds__(256) void ln_kernel(
    const float* __restrict__ x, const float* __restrict__ g,
    const float* __restrict__ be, _Float16* __restrict__ out) {
  __shared__ float red[256];
  const int tid = threadIdx.x;
  const size_t base = (size_t)blockIdx.x * Hc;
  float v0 = x[base + tid], v1 = x[base + tid + 256], v2 = x[base + tid + 512];
  red[tid] = v0 + v1 + v2;
  __syncthreads();
  for (int o = 128; o > 0; o >>= 1) {
    if (tid < o) red[tid] += red[tid + o];
    __syncthreads();
  }
  const float mean = red[0] * (1.0f / (float)Hc);
  __syncthreads();
  float d0 = v0 - mean, d1 = v1 - mean, d2 = v2 - mean;
  red[tid] = d0 * d0 + d1 * d1 + d2 * d2;
  __syncthreads();
  for (int o = 128; o > 0; o >>= 1) {
    if (tid < o) red[tid] += red[tid + o];
    __syncthreads();
  }
  const float inv = rsqrtf(red[0] * (1.0f / (float)Hc) + 1e-5f);
  out[base + tid]       = (_Float16)(d0 * inv * g[tid]       + be[tid]);
  out[base + tid + 256] = (_Float16)(d1 * inv * g[tid + 256] + be[tid + 256]);
  out[base + tid + 512] = (_Float16)(d2 * inv * g[tid + 512] + be[tid + 512]);
}

// ---------------------------------------------------------------------------
// WMMA GEMM: C(MxN) = A(MxK f16 row-major) * W^T(NxK f16 row-major)^T.
// Block tile 128x64, 8 waves, each wave a 32x32 tile (4 WMMA accs).
// B (=W^T rows n0..n0+63, 32 k's) staged by async DMA into double-buffered
// LDS; one barrier per K-step; A fragments b128 direct from global.
// Epilogue modes:
//   0: Q  -> (acc+bias)*scale -> q16[head][t][d]
//   1: K  ->  acc+bias        -> k16[head][t][d]
//   2: V  ->  acc+bias        -> vT16[head][d][t]   (pre-transposed)
//   3: acc+bias + residual    -> f32 out[m][n]      (proj / FFN2)
//   4: gelu(acc+bias)         -> f16 out[m][n]      (FFN1)
// ---------------------------------------------------------------------------
__global__ __launch_bounds__(256) void wmma_gemm(
    const _Float16* __restrict__ A, const _Float16* __restrict__ WT,
    const float* __restrict__ bias, const float* __restrict__ resid,
    void* __restrict__ out, int M, int K, int N, int mode, float scale) {
  __shared__ _Float16 bT[2][64 * 32];       // [buf][n_local][k_local]

  const int tid = threadIdx.x, lane = tid & 31, wave = tid >> 5;
  const int hsel = lane >> 4, l16 = lane & 15;
  const int mi = wave & 3, ni = wave >> 2;
  const int m0 = blockIdx.x * 128, n0 = blockIdx.y * 64;

  // per-thread async staging slot: one b128 (16B) per thread per tile
  const int nn = tid >> 2, part = (tid & 3) * 8;       // 8 halves = 16 bytes
  const _Float16* gsrc = WT + (size_t)(n0 + nn) * K + part;
  const unsigned lds0 = (unsigned)(size_t)&bT[0][nn * 32 + part];
  const unsigned lds1 = (unsigned)(size_t)&bT[1][nn * 32 + part];

  v8f acc[2][2] = {};
  const _Float16* Arow0 = A + (size_t)(m0 + mi * 32 + l16) * K;
  const _Float16* Arow1 = Arow0 + (size_t)16 * K;

  async_b128(lds0, gsrc);                   // prefetch tile 0
  const int NT = K >> 5;
  for (int it = 0; it < NT; ++it) {
    const int k0 = it << 5;

    // A fragments first: global b128 loads overlap the async wait
    Frag16 a0, a1;
#pragma unroll
    for (int j = 0; j < 8; ++j) {
      const int kf = k0 + koff(j, hsel);
      a0.u[j] = *(const unsigned int*)(Arow0 + kf);
      a1.u[j] = *(const unsigned int*)(Arow1 + kf);
    }

    wait_async0();
    __syncthreads();
    if (it + 1 < NT)                        // stage next tile into other buf
      async_b128((it & 1) ? lds0 : lds1, gsrc + k0 + 32);

    Frag16 fb0, fb1;
    const _Float16* b0p = &bT[it & 1][(ni * 32 + l16) * 32];
    const _Float16* b1p = b0p + 16 * 32;
#pragma unroll
    for (int j = 0; j < 8; ++j) {
      const int kf = koff(j, hsel);
      fb0.u[j] = *(const unsigned int*)(b0p + kf);
      fb1.u[j] = *(const unsigned int*)(b1p + kf);
    }
    acc[0][0] = __builtin_amdgcn_wmma_f32_16x16x32_f16(false, a0.h, false, fb0.h,
                                                       (short)0, acc[0][0], false, false);
    acc[0][1] = __builtin_amdgcn_wmma_f32_16x16x32_f16(false, a0.h, false, fb1.h,
                                                       (short)0, acc[0][1], false, false);
    acc[1][0] = __builtin_amdgcn_wmma_f32_16x16x32_f16(false, a1.h, false, fb0.h,
                                                       (short)0, acc[1][0], false, false);
    acc[1][1] = __builtin_amdgcn_wmma_f32_16x16x32_f16(false, a1.h, false, fb1.h,
                                                       (short)0, acc[1][1], false, false);
  }

  // ---- epilogue ----
#pragma unroll
  for (int ai = 0; ai < 2; ++ai) {
#pragma unroll
    for (int bj = 0; bj < 2; ++bj) {
#pragma unroll
      for (int r = 0; r < 8; ++r) {
        const int m = m0 + mi * 32 + ai * 16 + r + 8 * hsel;
        const int n = n0 + ni * 32 + bj * 16 + l16;
        float v = acc[ai][bj][r] + bias[n];
        if (mode <= 2) {
          const int t = m >> 2, bb = m & 3;        // token = t*B + b, B=4
          const int hh = n >> 6, d = n & 63;
          const int head = bb * NHc + hh;
          _Float16* o = (_Float16*)out;
          if (mode == 0) v *= scale;
          if (mode == 2)
            o[((size_t)head * HDc + d) * Tc + t] = (_Float16)v;   // V^T
          else
            o[((size_t)head * Tc + t) * HDc + d] = (_Float16)v;   // Q, K
        } else if (mode == 3) {
          ((float*)out)[(size_t)m * N + n] = resid[(size_t)m * N + n] + v;
        } else {
          ((_Float16*)out)[(size_t)m * N + n] = (_Float16)gelu_f(v);
        }
      }
    }
  }
}

// ---------------------------------------------------------------------------
// Attention: block = (head, 64-query tile). Q.K^T via WMMA, scores in LDS,
// fused bias/decay/mask + softmax, then P.V via WMMA, writing token-major
// f16 output so the out-projection is a plain row-major GEMM.
// Dynamic LDS: 64x512 f32 scores (128KB) + 64x512 f16 probs (64KB) = 192KB.
// ---------------------------------------------------------------------------
__global__ __launch_bounds__(256) void attn_kernel(
    const _Float16* __restrict__ q, const _Float16* __restrict__ k,
    const _Float16* __restrict__ vT, const float* __restrict__ abias,
    const float* __restrict__ decay, const int* __restrict__ am,
    _Float16* __restrict__ outTok) {
  extern __shared__ char smem[];
  float* S = (float*)smem;                           // [64][512]
  _Float16* P = (_Float16*)(smem + 64 * 512 * 4);    // [64][512]

  const int head = blockIdx.y;
  const int bb = head / NHc, hh = head % NHc;
  const int t0 = blockIdx.x * 64;
  const int tid = threadIdx.x, lane = tid & 31, wave = tid >> 5;
  const int hsel = lane >> 4, l16 = lane & 15;
  const int mi = wave & 3, ni0 = (wave >> 2) * 2;

  const _Float16* qrow = q + ((size_t)head * Tc + t0 + mi * 16 + l16) * HDc;

  // ---- phase 1: scores = Q.K^T, + bias, * decay, mask -> S ----
  for (int s0 = 0; s0 < Tc; s0 += 64) {
    v8f acc0 = {}, acc1 = {};
#pragma unroll
    for (int kk = 0; kk < HDc; kk += 32) {
      Frag16 a, fb0, fb1;
#pragma unroll
      for (int j = 0; j < 8; ++j)
        a.u[j] = *(const unsigned int*)(qrow + kk + koff(j, hsel));
      const _Float16* k0p =
          k + ((size_t)head * Tc + s0 + ni0 * 16 + l16) * HDc + kk;
      const _Float16* k1p = k0p + 16 * HDc;
#pragma unroll
      for (int j = 0; j < 8; ++j) {
        int kf = koff(j, hsel);
        fb0.u[j] = *(const unsigned int*)(k0p + kf);
        fb1.u[j] = *(const unsigned int*)(k1p + kf);
      }
      acc0 = __builtin_amdgcn_wmma_f32_16x16x32_f16(false, a.h, false, fb0.h,
                                                    (short)0, acc0, false, false);
      acc1 = __builtin_amdgcn_wmma_f32_16x16x32_f16(false, a.h, false, fb1.h,
                                                    (short)0, acc1, false, false);
    }
#pragma unroll
    for (int r = 0; r < 8; ++r) {
      const int ml = mi * 16 + r + 8 * hsel;
      const int tg = t0 + ml;
      const bool amt = am[bb * Tc + tg] != 0;
#pragma unroll
      for (int which = 0; which < 2; ++which) {
        const int sl = s0 + (ni0 + which) * 16 + l16;
        const size_t bi = ((size_t)head * Tc + tg) * Tc + sl;
        float v = ((which ? acc1[r] : acc0[r]) + abias[bi]) * decay[bi];
        const bool ams = am[bb * Tc + sl] != 0;
        const bool masked = (!(amt && ams)) || (tg >= 1 && sl == 0);
        S[ml * Tc + sl] = masked ? NEG_INF : v;
      }
    }
  }
  __syncthreads();

  // ---- phase 2: softmax over s (4 lanes per row) ----
  {
    const int row = tid >> 2, sub = tid & 3;
    float* Sr = S + row * Tc;
    _Float16* Pr = P + row * Tc;
    float mx = NEG_INF;
    for (int j = 0; j < 128; ++j) mx = fmaxf(mx, Sr[sub + 4 * j]);
    mx = fmaxf(mx, __shfl_xor(mx, 1, 4));
    mx = fmaxf(mx, __shfl_xor(mx, 2, 4));
    float sum = 0.0f;
    for (int j = 0; j < 128; ++j) {
      float v = Sr[sub + 4 * j];
      float e = (v == NEG_INF) ? 0.0f : __expf(v - mx);
      Sr[sub + 4 * j] = e;
      sum += e;
    }
    sum += __shfl_xor(sum, 1, 4);
    sum += __shfl_xor(sum, 2, 4);
    const float inv = (sum > 0.0f) ? 1.0f / sum : 0.0f;
    for (int j = 0; j < 128; ++j)
      Pr[sub + 4 * j] = (_Float16)(Sr[sub + 4 * j] * inv);
  }
  __syncthreads();

  // ---- phase 3: out = P.V  (A from LDS, B from pre-transposed V^T) ----
  {
    v8f acc0 = {}, acc1 = {};
    const _Float16* Prow = P + (mi * 16 + l16) * Tc;
    for (int s0 = 0; s0 < Tc; s0 += 32) {
      Frag16 a, fb0, fb1;
#pragma unroll
      for (int j = 0; j < 8; ++j)
        a.u[j] = *(const unsigned int*)(Prow + s0 + koff(j, hsel));
      const _Float16* v0p =
          vT + ((size_t)head * HDc + ni0 * 16 + l16) * Tc + s0;
      const _Float16* v1p = v0p + 16 * Tc;
#pragma unroll
      for (int j = 0; j < 8; ++j) {
        int kf = koff(j, hsel);
        fb0.u[j] = *(const unsigned int*)(v0p + kf);
        fb1.u[j] = *(const unsigned int*)(v1p + kf);
      }
      acc0 = __builtin_amdgcn_wmma_f32_16x16x32_f16(false, a.h, false, fb0.h,
                                                    (short)0, acc0, false, false);
      acc1 = __builtin_amdgcn_wmma_f32_16x16x32_f16(false, a.h, false, fb1.h,
                                                    (short)0, acc1, false, false);
    }
#pragma unroll
    for (int r = 0; r < 8; ++r) {
      const int tg = t0 + mi * 16 + r + 8 * hsel;
      const int token = tg * Bc + bb;
#pragma unroll
      for (int which = 0; which < 2; ++which) {
        const int d = (ni0 + which) * 16 + l16;
        outTok[(size_t)token * Hc + hh * HDc + d] =
            (_Float16)(which ? acc1[r] : acc0[r]);
      }
    }
  }
}

// ---------------------------------------------------------------------------
// Host-side orchestration.
// ---------------------------------------------------------------------------
extern "C" void kernel_launch(void* const* d_in, const int* in_sizes, int n_in,
                              void* d_out, int out_size, void* d_ws,
                              size_t ws_size, hipStream_t stream) {
  const float* x         = (const float*)d_in[0];
  const float* attn_bias = (const float*)d_in[1];
  const float* decay     = (const float*)d_in[2];
  const int*   amask     = (const int*)  d_in[3];
  const float* Wq  = (const float*)d_in[4];
  const float* bq  = (const float*)d_in[5];
  const float* Wk  = (const float*)d_in[6];
  const float* bk  = (const float*)d_in[7];
  const float* Wv  = (const float*)d_in[8];
  const float* bv  = (const float*)d_in[9];
  const float* Wo  = (const float*)d_in[10];
  const float* bo  = (const float*)d_in[11];
  const float* g1  = (const float*)d_in[12];
  const float* be1 = (const float*)d_in[13];
  const float* g2  = (const float*)d_in[14];
  const float* be2 = (const float*)d_in[15];
  const float* W1  = (const float*)d_in[16];
  const float* b1  = (const float*)d_in[17];
  const float* W2  = (const float*)d_in[18];
  const float* b2  = (const float*)d_in[19];

  // workspace layout
  char* ws = (char*)d_ws;
  size_t off = 0;
  auto take = [&](size_t bytes) {
    char* p = ws + off;
    off = (off + bytes + 255) & ~(size_t)255;
    return p;
  };
  const size_t SQ = (size_t)Hc * Hc;      // 768*768
  const size_t S1 = (size_t)Hc * FFc;     // 768*3072
  const size_t LSTRIDE = 4 * SQ + 2 * S1; // f16 elems per layer of W^T

  float*     xA      = (float*)    take((size_t)Mtok * Hc * 4);
  float*     xB      = (float*)    take((size_t)Mtok * Hc * 4);
  _Float16*  h16     = (_Float16*) take((size_t)Mtok * Hc * 2);
  _Float16*  q16     = (_Float16*) take((size_t)Bc * NHc * Tc * HDc * 2);
  _Float16*  k16     = (_Float16*) take((size_t)Bc * NHc * Tc * HDc * 2);
  _Float16*  vT16    = (_Float16*) take((size_t)Bc * NHc * Tc * HDc * 2);
  _Float16*  attnTok = (_Float16*) take((size_t)Mtok * Hc * 2);
  _Float16*  gg16    = (_Float16*) take((size_t)Mtok * FFc * 2);
  _Float16*  wT      = (_Float16*) take((size_t)Lc * LSTRIDE * 2);
  (void)ws_size; (void)in_sizes; (void)n_in; (void)out_size;

  const float SCALE = 0.125f;   // HD^-0.5
  const dim3 blk(256);
  const dim3 gH(Mtok / 128, Hc / 64);    // 16 x 12
  const dim3 gF(Mtok / 128, FFc / 64);   // 16 x 48
  const dim3 gA(Tc / 64, Bc * NHc);      // 8 x 48
  const size_t attn_lds = (size_t)64 * 512 * 4 + (size_t)64 * 512 * 2;

  // ---- one-time weight conversion: f32 KxN -> f16 NxK ----
  for (int l = 0; l < Lc; ++l) {
    _Float16* base = wT + (size_t)l * LSTRIDE;
    transpose_w<<<dim3(12, 12), blk, 0, stream>>>(Wq + l * SQ, base,          Hc, Hc);
    transpose_w<<<dim3(12, 12), blk, 0, stream>>>(Wk + l * SQ, base + SQ,     Hc, Hc);
    transpose_w<<<dim3(12, 12), blk, 0, stream>>>(Wv + l * SQ, base + 2 * SQ, Hc, Hc);
    transpose_w<<<dim3(12, 12), blk, 0, stream>>>(Wo + l * SQ, base + 3 * SQ, Hc, Hc);
    transpose_w<<<dim3(12, 48), blk, 0, stream>>>(W1 + l * S1, base + 4 * SQ, Hc, FFc);
    transpose_w<<<dim3(48, 12), blk, 0, stream>>>(W2 + l * S1, base + 4 * SQ + S1, FFc, Hc);
  }

  for (int l = 0; l < Lc; ++l) {
    const float* xin = (l == 0) ? x : xA;
    _Float16* base = wT + (size_t)l * LSTRIDE;
    _Float16 *WqT = base, *WkT = base + SQ, *WvT = base + 2 * SQ,
             *WoT = base + 3 * SQ, *W1T = base + 4 * SQ, *W2T = base + 4 * SQ + S1;

    ln_kernel<<<Mtok, blk, 0, stream>>>(xin, g1 + l * Hc, be1 + l * Hc, h16);

    wmma_gemm<<<gH, blk, 0, stream>>>(h16, WqT, bq + l * Hc, nullptr,
                                      q16, Mtok, Hc, Hc, 0, SCALE);
    wmma_gemm<<<gH, blk, 0, stream>>>(h16, WkT, bk + l * Hc, nullptr,
                                      k16, Mtok, Hc, Hc, 1, 1.0f);
    wmma_gemm<<<gH, blk, 0, stream>>>(h16, WvT, bv + l * Hc, nullptr,
                                      vT16, Mtok, Hc, Hc, 2, 1.0f);

    attn_kernel<<<gA, blk, attn_lds, stream>>>(q16, k16, vT16, attn_bias,
                                               decay, amask, attnTok);

    wmma_gemm<<<gH, blk, 0, stream>>>(attnTok, WoT, bo + l * Hc, xin,
                                      xB, Mtok, Hc, Hc, 3, 1.0f);

    ln_kernel<<<Mtok, blk, 0, stream>>>(xB, g2 + l * Hc, be2 + l * Hc, h16);

    wmma_gemm<<<gF, blk, 0, stream>>>(h16, W1T, b1 + (size_t)l * FFc,
                                      nullptr, gg16, Mtok, Hc, FFc, 4, 1.0f);

    float* xo = (l == Lc - 1) ? (float*)d_out : xA;
    wmma_gemm<<<gH, blk, 0, stream>>>(gg16, W2T, b2 + l * Hc, xB, xo,
                                      Mtok, FFc, Hc, 3, 1.0f);
  }
}
